// DropGIN_3633542332750
// MI455X (gfx1250) — matrix operand order repaired and checked by reading, not compiled, not run on hardware
//
#include <hip/hip_runtime.h>

#define N_NODES 20000
#define N_EDGES 320000
#define DIM     64
#define NLAYERS 4
#define NCLASS  10
#define NRUNS   8
#define NGRAPH  200

typedef __attribute__((ext_vector_type(16))) _Float16 v16h;
typedef __attribute__((ext_vector_type(8)))  float    v8f;

#if defined(__has_builtin)
#  if __has_builtin(__builtin_amdgcn_global_load_async_to_lds_b128)
#    define HAVE_ASYNC_LDS 1
#  else
#    define HAVE_ASYNC_LDS 0
#  endif
#  if __has_builtin(__builtin_amdgcn_s_wait_asynccnt)
#    define HAVE_WAIT_ASYNC 1
#  else
#    define HAVE_WAIT_ASYNC 0
#  endif
#else
#  define HAVE_ASYNC_LDS 0
#  define HAVE_WAIT_ASYNC 0
#endif

#if HAVE_ASYNC_LDS
typedef int v4i __attribute__((vector_size(16)));
typedef __attribute__((address_space(1))) v4i v4i_gbl;   // global (__device__) AS
typedef __attribute__((address_space(3))) v4i v4i_lds;   // shared (LDS) AS
#endif

// ---------------- utility ----------------
__global__ void zero_i32(int* __restrict__ p, int n) {
    int i = blockIdx.x * blockDim.x + threadIdx.x;
    if (i < n) p[i] = 0;
}

// Fold BN (eval) + linear bias into per-column scale/shift:
//   out = relu(z * s + t),  s = gamma*rsqrt(var+eps),  t = (b - mean)*s + beta
__global__ void st_prep(const float* __restrict__ g1, const float* __restrict__ be1,
                        const float* __restrict__ mn1, const float* __restrict__ vr1,
                        const float* __restrict__ bb1,
                        const float* __restrict__ g2, const float* __restrict__ be2,
                        const float* __restrict__ mn2, const float* __restrict__ vr2,
                        const float* __restrict__ bb2,
                        float* __restrict__ st) {
    int f = threadIdx.x;
    if (f >= DIM) return;
    float s1 = g1[f] * rsqrtf(vr1[f] + 1e-5f);
    st[f]           = s1;
    st[DIM + f]     = (bb1[f] - mn1[f]) * s1 + be1[f];
    float s2 = g2[f] * rsqrtf(vr2[f] + 1e-5f);
    st[2 * DIM + f] = s2;
    st[3 * DIM + f] = (bb2[f] - mn2[f]) * s2 + be2[f];
}

// Pack f32 row-major W[64][64] into the WMMA 16-bit B-matrix (32x16) lane layout,
// so each GEMM lane loads 16 contiguous halves (32B) per fragment.
// Fragment (nt, ks): lane n<16: element e -> W[ks*32 + e][nt*16 + n]
//                    lane n>=16: element e -> W[ks*32 + 16 + e][nt*16 + n - 16]
__global__ void pack_w(const float* __restrict__ W, _Float16* __restrict__ Bpk) {
    int tid = blockIdx.x * blockDim.x + threadIdx.x;
    if (tid >= 4096) return;
    int e    = tid & 15;
    int lane = (tid >> 4) & 31;
    int ks   = (tid >> 9) & 1;
    int nt   = tid >> 10;
    int K    = ks * 32 + ((lane >= 16) ? 16 : 0) + e;
    int Ncol = nt * 16 + (lane & 15);
    Bpk[tid] = (_Float16)W[K * DIM + Ncol];
}

// ---------------- CSR build (by destination) ----------------
__global__ void deg_count(const int* __restrict__ dst, int* __restrict__ deg) {
    int e = blockIdx.x * blockDim.x + threadIdx.x;
    if (e < N_EDGES) atomicAdd(&deg[dst[e]], 1);
}

__global__ void scan_serial(const int* __restrict__ deg, int* __restrict__ offs) {
    if (threadIdx.x == 0 && blockIdx.x == 0) {
        int acc = 0;
        for (int n = 0; n < N_NODES; ++n) { offs[n] = acc; acc += deg[n]; }
        offs[N_NODES] = acc;
    }
}

__global__ void csr_fill(const int* __restrict__ src, const int* __restrict__ dst,
                         const int* __restrict__ offs, int* __restrict__ cur,
                         int* __restrict__ csr) {
    int e = blockIdx.x * blockDim.x + threadIdx.x;
    if (e >= N_EDGES) return;
    int d = dst[e];
    int pos = offs[d] + atomicAdd(&cur[d], 1);
    csr[pos] = src[e];
}

// ---------------- h0 = drop_mask ? 0 : x, replicated over runs ----------------
__global__ void h0_build(const float* __restrict__ x, const unsigned char* __restrict__ mask,
                         float* __restrict__ h) {
    int tid = blockIdx.x * blockDim.x + threadIdx.x;
    if (tid >= NRUNS * N_NODES * DIM) return;
    int rn = tid >> 6;
    int f  = tid & 63;
    int n  = rn % N_NODES;
    h[tid] = mask[rn] ? 0.0f : x[n * DIM + f];
}

// ---------------- GIN aggregation: pre = h + sum_{src->n} h[src] ----------------
// One wave per (run, node); each lane carries 2 consecutive features (float2).
// CSR gather keeps everything as plain loads (L2-resident, no atomics).
__global__ void __launch_bounds__(256) gin_aggregate(
        const float* __restrict__ h, const int* __restrict__ offs,
        const int* __restrict__ csr, float* __restrict__ pre) {
    int wave = blockIdx.x * (blockDim.x >> 5) + (threadIdx.x >> 5);
    int lane = threadIdx.x & 31;
    if (wave >= NRUNS * N_NODES) return;          // wave-uniform
    int r = wave / N_NODES;
    int n = wave - r * N_NODES;
    const float* hr = h + (size_t)r * N_NODES * DIM;
    int f = lane << 1;
    float2 acc = *(const float2*)(hr + (size_t)n * DIM + f);   // self term (eps=0)
    int e0 = offs[n], e1 = offs[n + 1];
    for (int e = e0; e < e1; ++e) {
        int sn = csr[e];
        if (e + 1 < e1)  // prefetch next neighbor row -> global_prefetch_b8
            __builtin_prefetch(hr + (size_t)csr[e + 1] * DIM + f, 0, 1);
        float2 v = *(const float2*)(hr + (size_t)sn * DIM + f);
        acc.x += v.x; acc.y += v.y;
    }
    *(float2*)(pre + ((size_t)r * N_NODES + n) * DIM + f) = acc;
}

// ---------------- WMMA GEMM + fused BN/bias + ReLU ----------------
// Out[M,64] = relu((A[M,64] @ W[64,64]) * s + t)
// Block stages packed B (8 KB) into LDS once (async-to-LDS if available),
// each wave keeps all 8 B fragments + epilogue coefficients in registers and
// grid-strides over 16-row strips: inner loop = load A, cvt, 8x v_wmma, store.
__global__ void __launch_bounds__(256) gin_gemm_bn_relu(
        const float* __restrict__ A, const _Float16* __restrict__ Bpk,
        const float* __restrict__ s, const float* __restrict__ t,
        float* __restrict__ Out, int M) {
    __shared__ _Float16 Blds[4096];   // full 64x64 f16 weight in WMMA-B layout

    int tid = threadIdx.x;
    // ---- stage packed B into LDS once per block (32B per thread) ----
    {
        const _Float16* g = Bpk + (size_t)tid * 16;
        _Float16* l = Blds + tid * 16;
#if HAVE_ASYNC_LDS
        v4i_gbl* gp = (v4i_gbl*)(unsigned long long)g;
        v4i_lds* lp = (v4i_lds*)(unsigned)(unsigned long long)l;  // low 32 bits = LDS offset
        __builtin_amdgcn_global_load_async_to_lds_b128(gp,     lp,     0, 0);
        __builtin_amdgcn_global_load_async_to_lds_b128(gp + 1, lp + 1, 0, 0);
#if HAVE_WAIT_ASYNC
        __builtin_amdgcn_s_wait_asynccnt(0);
#else
        asm volatile("s_wait_asynccnt 0x0" ::: "memory");
#endif
#else
        *(float4*)l       = *(const float4*)g;
        *(float4*)(l + 8) = *(const float4*)(g + 8);
#endif
    }
    __syncthreads();

    int lane     = tid & 31;
    int half_sel = lane >> 4;
    int colbase  = lane & 15;
    int wave     = blockIdx.x * (blockDim.x >> 5) + (tid >> 5);
    int nwaves   = gridDim.x * (blockDim.x >> 5);

    // ---- B fragments: LDS -> registers, reused for every strip ----
    v16h bfrag[8];
#pragma unroll
    for (int fi = 0; fi < 8; ++fi)
        bfrag[fi] = *(const v16h*)(Blds + ((fi * 32 + lane) << 4));

    // ---- per-lane epilogue coefficients (4 output columns per lane) ----
    float sc[4], tc[4];
#pragma unroll
    for (int nt = 0; nt < 4; ++nt) {
        sc[nt] = s[nt * 16 + colbase];
        tc[nt] = t[nt * 16 + colbase];
    }

    int nstrips = M >> 4;
    for (int strip = wave; strip < nstrips; strip += nwaves) {
        int rowbase = strip << 4;
        const float* Arow = A + ((size_t)rowbase + colbase) * DIM;

        // A fragments (contiguous K order, matching packed B):
        // lanes 0-15:  elements 0..15 -> K=koff+0..15
        // lanes 16-31: elements 0..15 -> K=koff+16..31
        v16h afrag[2];
#pragma unroll
        for (int ks = 0; ks < 2; ++ks) {
            int cb = ks * 32 + half_sel * 16;
            const float4* p0 = (const float4*)(Arow + cb);
            const float4* p1 = (const float4*)(Arow + cb + 8);
            float4 x0 = p0[0], x1 = p0[1];
            float4 y0 = p1[0], y1 = p1[1];
            v16h a;
            a[0]  = (_Float16)x0.x; a[1]  = (_Float16)x0.y; a[2]  = (_Float16)x0.z; a[3]  = (_Float16)x0.w;
            a[4]  = (_Float16)x1.x; a[5]  = (_Float16)x1.y; a[6]  = (_Float16)x1.z; a[7]  = (_Float16)x1.w;
            a[8]  = (_Float16)y0.x; a[9]  = (_Float16)y0.y; a[10] = (_Float16)y0.z; a[11] = (_Float16)y0.w;
            a[12] = (_Float16)y1.x; a[13] = (_Float16)y1.y; a[14] = (_Float16)y1.z; a[15] = (_Float16)y1.w;
            afrag[ks] = a;
        }

        v8f zacc = {};
        v8f acc[4] = {zacc, zacc, zacc, zacc};
#pragma unroll
        for (int nt = 0; nt < 4; ++nt) {
#pragma unroll
            for (int ks = 0; ks < 2; ++ks) {
                acc[nt] = __builtin_amdgcn_wmma_f32_16x16x32_f16(
                    false, afrag[ks], false, bfrag[nt * 2 + ks], (short)0, acc[nt], false, false);
            }
        }

        // C/D layout: VGPR r: lanes 0-15 -> M=r, lanes 16-31 -> M=r+8; N = lane%16
#pragma unroll
        for (int nt = 0; nt < 4; ++nt) {
#pragma unroll
            for (int r = 0; r < 8; ++r) {
                int orow = rowbase + r + half_sel * 8;
                int col  = nt * 16 + colbase;
                float v = acc[nt][r] * sc[nt] + tc[nt];
                Out[(size_t)orow * DIM + col] = fmaxf(v, 0.0f);
            }
        }
    }
}

// ---------------- readout: mean over runs, segment-sum over batch ----------------
__global__ void pool_mean_scatter(const float* __restrict__ h, const int* __restrict__ batch,
                                  float* __restrict__ g) {
    int tid = blockIdx.x * blockDim.x + threadIdx.x;
    if (tid >= N_NODES * DIM) return;
    int n = tid >> 6, f = tid & 63;
    float ssum = 0.0f;
#pragma unroll
    for (int r = 0; r < NRUNS; ++r) ssum += h[((size_t)r * N_NODES + n) * DIM + f];
    atomicAdd(&g[batch[n] * DIM + f], ssum * (1.0f / NRUNS));
}

// ---------------- final: five tiny FCs + log_softmax ----------------
__global__ void logits_lsm(const float* __restrict__ gbuf,
                           const float* w0, const float* w1, const float* w2,
                           const float* w3, const float* w4,
                           const float* b0, const float* b1, const float* b2,
                           const float* b3, const float* b4,
                           float* __restrict__ out) {
    int gi = blockIdx.x * blockDim.x + threadIdx.x;
    if (gi >= NGRAPH) return;
    const float* ws[5] = {w0, w1, w2, w3, w4};
    const float* bs[5] = {b0, b1, b2, b3, b4};
    float l[NCLASS];
    for (int c = 0; c < NCLASS; ++c) l[c] = 0.0f;
    for (int i = 0; i < 5; ++i) {
        const float* gv = gbuf + ((size_t)i * NGRAPH + gi) * DIM;
        for (int c = 0; c < NCLASS; ++c) l[c] += bs[i][c];
        for (int f = 0; f < DIM; ++f) {
            float gf = gv[f];
            const float* wr = ws[i] + f * NCLASS;
            for (int c = 0; c < NCLASS; ++c) l[c] += gf * wr[c];
        }
    }
    float mx = l[0];
    for (int c = 1; c < NCLASS; ++c) mx = fmaxf(mx, l[c]);
    float se = 0.0f;
    for (int c = 0; c < NCLASS; ++c) se += __expf(l[c] - mx);
    float lse = mx + __logf(se);
    for (int c = 0; c < NCLASS; ++c) out[gi * NCLASS + c] = l[c] - lse;
}

// ---------------- host orchestration ----------------
// d_in layout (jax-style alphabetical dict flattening):
//   0:x 1:edge_index 2:batch 3:drop_mask
//   4..19   : bns[l]  -> beta, gamma, mean, var            (4 per layer)
//   20..51  : convs[l]-> bn1.{beta,gamma,mean,var}, lin1.{b,w}, lin2.{b,w}
//   52..61  : fcs[i]  -> b, w
extern "C" void kernel_launch(void* const* d_in, const int* in_sizes, int n_in,
                              void* d_out, int out_size, void* d_ws, size_t ws_size,
                              hipStream_t stream) {
    (void)in_sizes; (void)n_in; (void)out_size; (void)ws_size;

    const float* x            = (const float*)d_in[0];
    const int*   ei           = (const int*)d_in[1];
    const int*   src          = ei;
    const int*   dst          = ei + N_EDGES;
    const int*   batch        = (const int*)d_in[2];
    const unsigned char* mask = (const unsigned char*)d_in[3];

    // --- workspace carve (256B aligned regions) ---
    char* ws = (char*)d_ws;
    size_t cur_off = 0;
    auto carve = [&](size_t bytes) -> char* {
        char* p = ws + cur_off;
        cur_off = (cur_off + bytes + 255) & ~(size_t)255;
        return p;
    };
    float*    P0   = (float*)carve((size_t)NRUNS * N_NODES * DIM * 4);   // 41 MB
    float*    P1   = (float*)carve((size_t)NRUNS * N_NODES * DIM * 4);   // 41 MB
    int*      deg  = (int*)carve((size_t)N_NODES * 4);
    int*      offs = (int*)carve((size_t)(N_NODES + 1) * 4);
    int*      curp = (int*)carve((size_t)N_NODES * 4);
    int*      csr  = (int*)carve((size_t)N_EDGES * 4);
    _Float16* wpk  = (_Float16*)carve((size_t)8 * 4096 * 2);
    float*    stv  = (float*)carve((size_t)NLAYERS * 4 * DIM * 4);
    float*    gbuf = (float*)carve((size_t)5 * NGRAPH * DIM * 4);

    // --- per-call zeroing (deterministic) ---
    zero_i32<<<(N_NODES + 255) / 256, 256, 0, stream>>>(deg, N_NODES);
    zero_i32<<<(N_NODES + 255) / 256, 256, 0, stream>>>(curp, N_NODES);
    zero_i32<<<(5 * NGRAPH * DIM + 255) / 256, 256, 0, stream>>>((int*)gbuf, 5 * NGRAPH * DIM);

    // --- parameter prep: packed f16 weights + folded BN affines ---
    for (int l = 0; l < NLAYERS; ++l) {
        const float* bn1_beta  = (const float*)d_in[20 + 8 * l + 0];
        const float* bn1_gamma = (const float*)d_in[20 + 8 * l + 1];
        const float* bn1_mean  = (const float*)d_in[20 + 8 * l + 2];
        const float* bn1_var   = (const float*)d_in[20 + 8 * l + 3];
        const float* lin1_b    = (const float*)d_in[20 + 8 * l + 4];
        const float* lin1_w    = (const float*)d_in[20 + 8 * l + 5];
        const float* lin2_b    = (const float*)d_in[20 + 8 * l + 6];
        const float* lin2_w    = (const float*)d_in[20 + 8 * l + 7];
        const float* bn_beta   = (const float*)d_in[4 + 4 * l + 0];
        const float* bn_gamma  = (const float*)d_in[4 + 4 * l + 1];
        const float* bn_mean   = (const float*)d_in[4 + 4 * l + 2];
        const float* bn_var    = (const float*)d_in[4 + 4 * l + 3];
        st_prep<<<1, 64, 0, stream>>>(bn1_gamma, bn1_beta, bn1_mean, bn1_var, lin1_b,
                                      bn_gamma, bn_beta, bn_mean, bn_var, lin2_b,
                                      stv + (size_t)l * 4 * DIM);
        pack_w<<<16, 256, 0, stream>>>(lin1_w, wpk + (size_t)(2 * l) * 4096);
        pack_w<<<16, 256, 0, stream>>>(lin2_w, wpk + (size_t)(2 * l + 1) * 4096);
    }

    // --- CSR by destination (turns scatter-atomics into L2-resident gathers) ---
    deg_count<<<(N_EDGES + 255) / 256, 256, 0, stream>>>(dst, deg);
    scan_serial<<<1, 1, 0, stream>>>(deg, offs);
    csr_fill<<<(N_EDGES + 255) / 256, 256, 0, stream>>>(src, dst, offs, curp, csr);

    // --- h0 + readout of layer-0 ---
    h0_build<<<(NRUNS * N_NODES * DIM + 255) / 256, 256, 0, stream>>>(x, mask, P0);
    float* h = P0;
    float* t = P1;
    pool_mean_scatter<<<(N_NODES * DIM + 255) / 256, 256, 0, stream>>>(h, batch, gbuf);

    const int M = NRUNS * N_NODES;          // 160000 rows
    const int gemm_blocks = 320;            // 2560 waves grid-striding 10000 strips
    const int agg_blocks  = (M + 7) / 8;    // one wave per (run,node)

    for (int l = 0; l < NLAYERS; ++l) {
        float* stl = stv + (size_t)l * 4 * DIM;
        gin_aggregate<<<agg_blocks, 256, 0, stream>>>(h, offs, csr, t);
        gin_gemm_bn_relu<<<gemm_blocks, 256, 0, stream>>>(
            t, wpk + (size_t)(2 * l) * 4096, stl, stl + DIM, h, M);
        gin_gemm_bn_relu<<<gemm_blocks, 256, 0, stream>>>(
            h, wpk + (size_t)(2 * l + 1) * 4096, stl + 2 * DIM, stl + 3 * DIM, t, M);
        pool_mean_scatter<<<(N_NODES * DIM + 255) / 256, 256, 0, stream>>>(
            t, batch, gbuf + (size_t)(l + 1) * NGRAPH * DIM);
        float* tmp = h; h = t; t = tmp;
    }

    logits_lsm<<<1, 256, 0, stream>>>(gbuf,
        (const float*)d_in[53], (const float*)d_in[55], (const float*)d_in[57],
        (const float*)d_in[59], (const float*)d_in[61],
        (const float*)d_in[52], (const float*)d_in[54], (const float*)d_in[56],
        (const float*)d_in[58], (const float*)d_in[60],
        (float*)d_out);
}
